// SubwordRep_53807350284699
// MI455X (gfx1250) — compile-verified, matrix operand order checked by reading, not run on hardware
//
#include <hip/hip_runtime.h>
#include <cstdint>
#include <cstddef>

// ---------------------------------------------------------------------------
// Problem dimensions (fixed by the reference)
// ---------------------------------------------------------------------------
constexpr int Bb   = 128;           // batch
constexpr int Ss   = 256;           // sequence length
constexpr int Ee   = 256;           // embedding dim
constexpr int Hh   = 256;           // hidden dim
constexpr int G4   = 4 * Hh;        // gate columns (i,f,g,o)
constexpr int KCAT = Ee + Hh;       // concatenated [x;h] reduction dim

typedef __bf16 bf16;
typedef __attribute__((ext_vector_type(16))) bf16  v16bf;
typedef __attribute__((ext_vector_type(8)))  bf16  v8bf;
typedef __attribute__((ext_vector_type(8)))  float v8f;

struct bf16x16 { v8bf lo, hi; };
__device__ __forceinline__ v16bf concat16(v8bf a, v8bf b) {
  bf16x16 t{a, b};
  return __builtin_bit_cast(v16bf, t);
}

__device__ __forceinline__ float sigm(float x) { return 1.0f / (1.0f + __expf(-x)); }

// ---------------------------------------------------------------------------
// K0: lengths, stable descending sort -> perm; out_len (perm space) -> perm2
// ---------------------------------------------------------------------------
__global__ void k_perm(const int* __restrict__ ids, const int* __restrict__ fm,
                       int* __restrict__ lengths, int* __restrict__ perm,
                       int* __restrict__ out_len, int* __restrict__ perm2) {
  const int b = threadIdx.x;
  int len = 0;
  for (int s = 0; s < Ss; ++s) len += (ids[b * Ss + s] != 0);
  lengths[b] = len;
  __syncthreads();
  int r = 0;
  for (int j = 0; j < Bb; ++j) {
    const int lj = lengths[j];
    r += (lj > len) || (lj == len && j < b);       // stable argsort(-len)
  }
  perm[r] = b;
  __syncthreads();
  const int orig = perm[b];
  int cnt = 0;
  for (int s = 0; s < Ss; ++s) cnt += (fm[orig * Ss + s] != 0);
  out_len[b] = cnt;
  __syncthreads();
  int r2 = 0;
  for (int j = 0; j < Bb; ++j) {
    const int cj = out_len[j];
    r2 += (cj > cnt) || (cj == cnt && j < b);      // stable argsort(-out_len)
  }
  perm2[r2] = b;
}

// ---------------------------------------------------------------------------
// K1: pack [Wih ; Whh] rows into bf16 Wcat[dir][n][KCAT] (L2-resident, 2MB)
// ---------------------------------------------------------------------------
__global__ void k_weights(const float* __restrict__ Wih_f, const float* __restrict__ Whh_f,
                          const float* __restrict__ Wih_b, const float* __restrict__ Whh_b,
                          bf16* __restrict__ Wcat) {
  const int n = blockIdx.x, dir = blockIdx.y, e = threadIdx.x;
  const float* Wih = dir ? Wih_b : Wih_f;
  const float* Whh = dir ? Whh_b : Whh_f;
  bf16* dst = Wcat + ((size_t)dir * G4 + n) * KCAT;
  dst[e]      = (bf16)Wih[(size_t)n * Ee + e];
  dst[Ee + e] = (bf16)Whh[(size_t)n * Hh + e];
}

// ---------------------------------------------------------------------------
// K2: embedding gather in perm order -> bf16 X[b][s][e]; validity mask
// ---------------------------------------------------------------------------
__global__ void k_embed(const int* __restrict__ ids, const int* __restrict__ perm,
                        const float* __restrict__ embW,
                        bf16* __restrict__ X, float* __restrict__ maskf) {
  const int s = blockIdx.x, b = blockIdx.y, e = threadIdx.x;
  const int orig = perm[b];
  const int id = ids[orig * Ss + s];
  X[(((size_t)b) * Ss + s) * Ee + e] = (bf16)embW[(size_t)id * Ee + e];
  if (e == 0) maskf[b * Ss + s] = (id != 0) ? 1.0f : 0.0f;
}

// ---------------------------------------------------------------------------
// K3: fused bidirectional LSTM recurrence.
//     One WG per 16 batch rows per direction; 8 waves, each owning a 32-column
//     j-slice of ALL four gate groups so the i/f/g/o WMMA fragments for the
//     same (m,j) land in the same lane+register (zero cross-lane cell math).
//     Weights stream from L2 every step (2MB total, L2-resident); the memory
//     clobber + unroll(1) keep the loads IN the loop so nothing spills.
// ---------------------------------------------------------------------------
constexpr int XH_LD = KCAT + 8;     // pad row stride to dodge LDS bank repeats

__global__ __launch_bounds__(256) void
k_lstm(const bf16* __restrict__ X, const float* __restrict__ maskf,
       const bf16* __restrict__ Wcat, const float* __restrict__ b_f,
       const float* __restrict__ b_b, float* __restrict__ out_seq) {
  __shared__ __align__(16) bf16 xh[16][XH_LD];   // [m][ x(0..255) | h(256..511) ]
  __shared__ float mrow[16];

  const int tid  = threadIdx.x;
  const int lane = tid & 31;
  const int wv   = tid >> 5;          // 8 waves
  const int hi   = lane >> 4;         // fragment half
  const int ln   = lane & 15;
  const int b0   = blockIdx.x * 16;   // batch-row slice (perm space)
  const int dir  = blockIdx.y;

  const bf16*  W    = Wcat + (size_t)dir * G4 * KCAT;
  const float* bias = dir ? b_b : b_f;
  float* outp = out_seq + (size_t)dir * Bb * Ss * Hh;

  // h := 0
  for (int i = tid; i < 16 * Hh; i += 256) {
    xh[i >> 8][Ee + (i & 255)] = (bf16)0.0f;
  }

  const int jbase = wv * 32;          // this wave's gate-column slice

  v8f c[2];                           // cell state, C-fragment layout
  #pragma unroll
  for (int s = 0; s < 2; ++s)
    #pragma unroll
    for (int r = 0; r < 8; ++r) c[s][r] = 0.0f;

  float bia[4][2];                    // per-lane gate biases
  #pragma unroll
  for (int g = 0; g < 4; ++g)
    #pragma unroll
    for (int s = 0; s < 2; ++s)
      bia[g][s] = bias[g * Hh + jbase + s * 16 + ln];

  const int koff0 = hi << 3;          // A/B fragment K sub-offset (0 or 8)
  // one per-lane weight base; per-(g,s) row deltas are compile-time constants
  // that fold into the 24-bit instruction offset of global_load_b128
  const bf16* wbase = W + (size_t)(jbase + ln) * KCAT + koff0;

  #pragma unroll 1
  for (int t = 0; t < Ss; ++t) {
    // forbid hoisting any loads (esp. L2-resident weights) across time steps
    asm volatile("" ::: "memory");

    const int tr = dir ? (Ss - 1 - t) : t;
    __syncthreads();                  // prev step's xh reads complete

    // stage x_t tile (16 x 256 bf16) + mask; prefetch next step's x row
    {
      const int m = tid >> 4, chunk = tid & 15;
      const bf16* xr = X + (((size_t)(b0 + m)) * Ss + tr) * Ee + chunk * 16;
      const v8bf x0 = *reinterpret_cast<const v8bf*>(xr);
      const v8bf x1 = *reinterpret_cast<const v8bf*>(xr + 8);
      *reinterpret_cast<v8bf*>(&xh[m][chunk * 16])     = x0;
      *reinterpret_cast<v8bf*>(&xh[m][chunk * 16 + 8]) = x1;
      if (t + 1 < Ss) {
        const int tn = dir ? (tr - 1) : (tr + 1);
        __builtin_prefetch(X + (((size_t)(b0 + m)) * Ss + tn) * Ee + chunk * 16, 0, 1);
      }
      if (tid < 16) mrow[tid] = maskf[(b0 + tid) * Ss + tr];
    }
    __syncthreads();                  // xh = [x_t ; h_{t-1}] ready

    // ---- gates = [x;h] @ Wcat^T + b : 8 N-tiles x 16 K-steps of bf16 WMMA
    v8f acc[4][2];
    #pragma unroll
    for (int g = 0; g < 4; ++g)
      #pragma unroll
      for (int s = 0; s < 2; ++s) {
        #pragma unroll
        for (int r = 0; r < 8; ++r) acc[g][s][r] = bia[g][s];
      }

    #pragma unroll 1                  // keep a real loop: bounds VGPR pressure
    for (int kk = 0; kk < KCAT / 32; ++kk) {
      const int k0 = kk * 32 + koff0;
      // A fragment from LDS (ISA 16-bit 16x32 layout), shared by all 8 tiles
      const v8bf a0 = *reinterpret_cast<const v8bf*>(&xh[ln][k0]);
      const v8bf a1 = *reinterpret_cast<const v8bf*>(&xh[ln][k0 + 16]);
      const v16bf A = concat16(a0, a1);

      const bf16* wk = wbase + kk * 32;
      #pragma unroll
      for (int g = 0; g < 4; ++g) {
        #pragma unroll
        for (int s = 0; s < 2; ++s) {
          const bf16* wr = wk + (size_t)(g * Hh + s * 16) * KCAT;  // const delta
          const v8bf w0 = *reinterpret_cast<const v8bf*>(wr);
          const v8bf w1 = *reinterpret_cast<const v8bf*>(wr + 16);
          const v16bf Bm = concat16(w0, w1);
          acc[g][s] = __builtin_amdgcn_wmma_f32_16x16x32_bf16(
              false, A, false, Bm, (short)0, acc[g][s], false, false);
        }
      }
    }
    __syncthreads();                  // all WMMA reads of h done

    // ---- LSTM cell, purely fragment-local (no cross-lane traffic) ----
    #pragma unroll
    for (int s = 0; s < 2; ++s) {
      const int j = jbase + s * 16 + ln;
      #pragma unroll
      for (int r = 0; r < 8; ++r) {
        const int   m  = r + hi * 8;
        const float mt = mrow[m];
        const float ig = sigm(acc[0][s][r]);
        const float fg = sigm(acc[1][s][r]);
        const float gg = tanhf(acc[2][s][r]);
        const float og = sigm(acc[3][s][r]);
        const float cn = fg * c[s][r] + ig * gg;
        const float hn = og * tanhf(cn);
        const float hold = (float)xh[m][Ee + j];
        const float cu = (mt > 0.0f) ? cn : c[s][r];
        const float hu = (mt > 0.0f) ? hn : hold;
        c[s][r] = cu;
        xh[m][Ee + j] = (bf16)hu;     // h feedback for step t+1
        outp[(((size_t)(b0 + m)) * Ss + tr) * Hh + j] = hu * mt;
      }
    }
  }
}

// ---------------------------------------------------------------------------
// K4: select-pad (compact masked positions), perm2 reorder, fwd||bwd concat,
//     zero the padding, and emit out_len tail. Writes EVERY output element.
// ---------------------------------------------------------------------------
__global__ __launch_bounds__(256) void
k_select(const int* __restrict__ perm, const int* __restrict__ perm2,
         const int* __restrict__ fm, const int* __restrict__ bm,
         const int* __restrict__ out_len, const float* __restrict__ out_seq,
         float* __restrict__ out) {
  __shared__ int gf[Ss], gb[Ss];
  __shared__ int cnt_s;
  const int b2  = blockIdx.x;
  const int row = perm2[b2];                    // perm-space row
  const int orig = perm[row];                   // original row (for masks)
  if (threadIdx.x == 0) {
    for (int s = 0; s < Ss; ++s) { gf[s] = 0; gb[s] = 0; }
    int cf = 0;
    for (int s = 0; s < Ss; ++s) if (fm[orig * Ss + s]) gf[cf++] = s;
    cnt_s = cf;
    int cb = 0;
    for (int s = 0; s < Ss; ++s) if (bm[orig * Ss + s]) gb[cb++] = s;
    out[(size_t)Bb * Ss * 2 * Hh + b2] = (float)out_len[row];   // tuple tail
  }
  __syncthreads();
  const int cnt = cnt_s;
  const float* fsrc = out_seq + ((size_t)0 * Bb + row) * Ss * Hh;
  const float* bsrc = out_seq + ((size_t)1 * Bb + row) * Ss * Hh;
  const int e = threadIdx.x;
  for (int w = 0; w < Ss; ++w) {
    float fv = 0.0f, bv = 0.0f;
    if (w < cnt) {
      fv = fsrc[(size_t)gf[w] * Hh + e];
      bv = bsrc[(size_t)gb[w] * Hh + e];
    }
    const size_t o = (((size_t)b2) * Ss + w) * (2 * Hh);
    out[o + e]      = fv;
    out[o + Hh + e] = bv;
  }
}

// ---------------------------------------------------------------------------
// Host side
// ---------------------------------------------------------------------------
extern "C" void kernel_launch(void* const* d_in, const int* in_sizes, int n_in,
                              void* d_out, int out_size, void* d_ws, size_t ws_size,
                              hipStream_t stream) {
  (void)in_sizes; (void)n_in; (void)out_size; (void)ws_size;
  const int*   sw_ids = (const int*)d_in[0];
  const int*   fmasks = (const int*)d_in[1];
  const int*   bmasks = (const int*)d_in[2];
  const float* emb_W  = (const float*)d_in[3];
  const float* W_ih_f = (const float*)d_in[4];
  const float* W_hh_f = (const float*)d_in[5];
  const float* b_f    = (const float*)d_in[6];
  const float* W_ih_b = (const float*)d_in[7];
  const float* W_hh_b = (const float*)d_in[8];
  const float* b_b    = (const float*)d_in[9];
  float* out = (float*)d_out;

  char* ws = (char*)d_ws;
  size_t off = 0;
  auto alloc = [&](size_t bytes) -> void* {
    off = (off + 255) & ~(size_t)255;
    void* p = ws + off;
    off += bytes;
    return p;
  };
  int*   lengths  = (int*)  alloc((size_t)Bb * 4);
  int*   perm     = (int*)  alloc((size_t)Bb * 4);
  int*   out_len  = (int*)  alloc((size_t)Bb * 4);
  int*   perm2    = (int*)  alloc((size_t)Bb * 4);
  bf16*  Wcat     = (bf16*) alloc((size_t)2 * G4 * KCAT * sizeof(bf16));     // 2 MB
  bf16*  X        = (bf16*) alloc((size_t)Bb * Ss * Ee * sizeof(bf16));      // 16 MB
  float* maskf    = (float*)alloc((size_t)Bb * Ss * sizeof(float));          // 128 KB
  float* out_seq  = (float*)alloc((size_t)2 * Bb * Ss * Hh * sizeof(float)); // 64 MB

  hipLaunchKernelGGL(k_perm, dim3(1), dim3(Bb), 0, stream,
                     sw_ids, fmasks, lengths, perm, out_len, perm2);
  hipLaunchKernelGGL(k_weights, dim3(G4, 2), dim3(Ee), 0, stream,
                     W_ih_f, W_hh_f, W_ih_b, W_hh_b, Wcat);
  hipLaunchKernelGGL(k_embed, dim3(Ss, Bb), dim3(Ee), 0, stream,
                     sw_ids, perm, emb_W, X, maskf);
  hipLaunchKernelGGL(k_lstm, dim3(Bb / 16, 2), dim3(256), 0, stream,
                     X, maskf, Wcat, b_f, b_b, out_seq);
  hipLaunchKernelGGL(k_select, dim3(Bb), dim3(256), 0, stream,
                     perm, perm2, fmasks, bmasks, out_len, out_seq, out);
}
